// LSTMCellLayerNorm_84696755077791
// MI455X (gfx1250) — matrix operand order repaired and checked by reading, compiled, not run on hardware
//
#include <hip/hip_runtime.h>
#include <hip/hip_bf16.h>

// ---------------------------------------------------------------------------
// LSTM cell + LayerNorm for MI455X (gfx1250), wave32 + WMMA + async LDS copy.
//
// preact = [x|h] (2048x2048 f16) @ [W_ih|W_hh]^T (4096x2048 f16), f32 accum.
// Compute-bound (AI ~460 FLOP/byte) -> v_wmma_f32_16x16x32_f16 path.
// Tile pipeline: GLOBAL_LOAD_ASYNC_TO_LDS_B128 double buffer (ASYNCcnt).
// ---------------------------------------------------------------------------

typedef __attribute__((ext_vector_type(16))) _Float16 v16h;
typedef __attribute__((ext_vector_type(8)))  _Float16 v8h;
typedef __attribute__((ext_vector_type(4)))  _Float16 v4h;
typedef __attribute__((ext_vector_type(8)))  float    v8f;

constexpr int HID   = 1024;   // hidden size
constexpr int BATCH = 2048;
constexpr int KTOT  = 2048;   // concatenated K = 2*HID
constexpr int NTOT  = 4096;   // 4*HID gate columns
constexpr float EPSV = 1e-5f;

constexpr int BM = 128, BN = 256, BK = 32;
constexpr int LDSTR = 40;     // padded f16 row stride in LDS (16B-aligned)

// ---- CDNA5 async global->LDS copy (ASYNCcnt-tracked) -----------------------
__device__ __forceinline__ void async_copy_b128(void* lds, const void* gsrc)
{
    // LDS byte address = low 32 bits of the generic shared-aperture pointer.
    unsigned l = (unsigned)(uintptr_t)lds;
    asm volatile("global_load_async_to_lds_b128 %0, %1, off"
                 :: "v"(l), "v"(gsrc) : "memory");
}
__device__ __forceinline__ void wait_async0()
{
    asm volatile("s_wait_asynccnt 0x0" ::: "memory");
}

// ---- kernel 1: f32 -> f16 convert with K-concatenation ---------------------
__global__ __launch_bounds__(256)
void pack_f16_kernel(const float* __restrict__ src, _Float16* __restrict__ dst,
                     int cols, int dst_stride, int dst_off, int total4)
{
    int idx = blockIdx.x * blockDim.x + threadIdx.x;
    if (idx >= total4) return;
    int e = idx * 4;
    int r = e / cols;
    int c = e - r * cols;
    const float4 f = *(const float4*)(src + e);
    v4h o;
    o.x = (_Float16)f.x; o.y = (_Float16)f.y;
    o.z = (_Float16)f.z; o.w = (_Float16)f.w;
    *(v4h*)(dst + (size_t)r * dst_stride + dst_off + c) = o;
}

// ---- kernel 2: tiled WMMA GEMM  C[BATCH x NTOT] = A @ B^T ------------------
// A = XH   [BATCH x KTOT] f16 row-major (K contiguous)
// B = Wcat [NTOT  x KTOT] f16 row-major (K contiguous)  -> perfect NT layout
// Block tile 128x256, 8 waves each owning a 64x64 sub-tile (16 WMMA / BK=32).
__global__ __launch_bounds__(256)
void gemm_wmma_kernel(const _Float16* __restrict__ A,
                      const _Float16* __restrict__ Bm,
                      float* __restrict__ C)
{
    __shared__ _Float16 As[2][BM][LDSTR];
    __shared__ _Float16 Bs[2][BN][LDSTR];

    const int tid  = threadIdx.x;
    const int lane = tid & 31;
    const int wave = tid >> 5;          // 8 waves
    const int wm   = wave >> 2;         // 2 row-bands of 64
    const int wn   = wave & 3;          // 4 col-bands of 64
    const int bm   = blockIdx.y * BM;
    const int bn   = blockIdx.x * BN;

    v8f acc[4][4] = {};                 // 16 x 16x16 f32 accumulators per wave

    union F16x16 { v16h v; v8h h[2]; };

    // async tile loader: 16B (8 x f16) chunks; A: 512 chunks, B: 1024 chunks
    auto load_tiles = [&](int b, int kt) {
        #pragma unroll
        for (int i = 0; i < 2; ++i) {
            int c4  = tid + i * 256;          // 0..511
            int row = c4 >> 2;                // 0..127
            int col = (c4 & 3) * 8;           // 0,8,16,24
            async_copy_b128(&As[b][row][col],
                            A + (size_t)(bm + row) * KTOT + kt + col);
        }
        #pragma unroll
        for (int i = 0; i < 4; ++i) {
            int c4  = tid + i * 256;          // 0..1023
            int row = c4 >> 2;                // 0..255
            int col = (c4 & 3) * 8;
            async_copy_b128(&Bs[b][row][col],
                            Bm + (size_t)(bn + row) * KTOT + kt + col);
        }
    };

    int buf = 0;
    load_tiles(0, 0);
    wait_async0();
    __syncthreads();

    for (int kt = 0; kt < KTOT; kt += BK) {
        if (kt + BK < KTOT)
            load_tiles(buf ^ 1, kt + BK);     // prefetch next tile (async)

        // A fragment (16x32 f16): lanes 0-15 row m hold K 0..7 & 16..23,
        // lanes 16-31 hold K 8..15 & 24..31 (ISA 7.12.2 layout).
        F16x16 af[4], bf[4];
        const int frow = lane & 15;
        const int akb  = (lane < 16) ? 0 : 8;
        #pragma unroll
        for (int mi = 0; mi < 4; ++mi) {
            int m = wm * 64 + mi * 16 + frow;
            af[mi].h[0] = *(const v8h*)&As[buf][m][akb];
            af[mi].h[1] = *(const v8h*)&As[buf][m][akb + 16];
        }
        // B fragment (32x16): lanes 0-15 col n hold K 0..7 & 16..23, lanes 16-31 K 8..15 & 24..31.
        const int bkb = (lane < 16) ? 0 : 8;
        #pragma unroll
        for (int ni = 0; ni < 4; ++ni) {
            int n = wn * 64 + ni * 16 + frow;
            bf[ni].h[0] = *(const v8h*)&Bs[buf][n][bkb];
            bf[ni].h[1] = *(const v8h*)&Bs[buf][n][bkb + 16];
        }

        #pragma unroll
        for (int mi = 0; mi < 4; ++mi)
            #pragma unroll
            for (int ni = 0; ni < 4; ++ni)
                acc[mi][ni] = __builtin_amdgcn_wmma_f32_16x16x32_f16(
                    false, af[mi].v, false, bf[ni].v,
                    (short)0, acc[mi][ni], false, false);

        wait_async0();      // next tile fully landed in LDS
        __syncthreads();    // includes DS-cnt wait: all reads of `buf` done
        buf ^= 1;
    }

    // C/D layout: VGPR r, lanes 0-15 -> M=r, lanes 16-31 -> M=r+8; N = lane%16.
    const int cn   = lane & 15;
    const int mofs = (lane < 16) ? 0 : 8;
    #pragma unroll
    for (int mi = 0; mi < 4; ++mi)
        #pragma unroll
        for (int ni = 0; ni < 4; ++ni) {
            int col = bn + wn * 64 + ni * 16 + cn;
            #pragma unroll
            for (int r = 0; r < 8; ++r) {
                int row = bm + wm * 64 + mi * 16 + r + mofs;
                C[(size_t)row * NTOT + col] = acc[mi][ni][r];
            }
        }
}

// ---- kernel 3: gates + cell update + LayerNorm + h_t -----------------------
__global__ __launch_bounds__(256)
void lstm_ln_kernel(const float* __restrict__ preact,
                    const float* __restrict__ c_in,
                    const float* __restrict__ b_ih, const float* __restrict__ b_hh,
                    const float* __restrict__ ln_w, const float* __restrict__ ln_b,
                    float* __restrict__ h_out, float* __restrict__ c_out)
{
    const int b   = blockIdx.x;
    const int tid = threadIdx.x;
    const float* row = preact + (size_t)b * NTOT;

    float ct[4], ot[4];
    float sum = 0.f, sq = 0.f;
    #pragma unroll
    for (int u = 0; u < 4; ++u) {
        int j = tid + u * 256;
        float pi = row[j]           + b_ih[j]           + b_hh[j];
        float pf = row[HID + j]     + b_ih[HID + j]     + b_hh[HID + j];
        float po = row[2*HID + j]   + b_ih[2*HID + j]   + b_hh[2*HID + j];
        float pg = row[3*HID + j]   + b_ih[3*HID + j]   + b_hh[3*HID + j];
        float it = 1.f / (1.f + __expf(-pi));
        float ft = 1.f / (1.f + __expf(-pf));
        ot[u]    = 1.f / (1.f + __expf(-po));
        float gt = tanhf(pg);
        float cv = c_in[(size_t)b * HID + j] * ft + it * gt;
        ct[u] = cv;
        sum += cv;
        sq  += cv * cv;
    }

    // wave32 butterfly + cross-wave LDS reduction (8 waves / 256 threads)
    #pragma unroll
    for (int off = 16; off > 0; off >>= 1) {
        sum += __shfl_down(sum, off, 32);
        sq  += __shfl_down(sq,  off, 32);
    }
    __shared__ float ssum[8], ssq[8];
    if ((tid & 31) == 0) { ssum[tid >> 5] = sum; ssq[tid >> 5] = sq; }
    __syncthreads();
    float ts = 0.f, tq = 0.f;
    #pragma unroll
    for (int w = 0; w < 8; ++w) { ts += ssum[w]; tq += ssq[w]; }

    const float mu   = ts * (1.f / HID);
    const float var  = tq * (1.f / HID) - mu * mu;
    const float rstd = rsqrtf(var + EPSV);

    #pragma unroll
    for (int u = 0; u < 4; ++u) {
        int j = tid + u * 256;
        float cnorm = (ct[u] - mu) * rstd * ln_w[j] + ln_b[j];
        c_out[(size_t)b * HID + j] = cnorm;
        h_out[(size_t)b * HID + j] = ot[u] * tanhf(cnorm);
    }
}

// ---------------------------------------------------------------------------
extern "C" void kernel_launch(void* const* d_in, const int* in_sizes, int n_in,
                              void* d_out, int out_size, void* d_ws, size_t ws_size,
                              hipStream_t stream)
{
    const float* x    = (const float*)d_in[0];
    const float* h    = (const float*)d_in[1];
    const float* c    = (const float*)d_in[2];
    const float* W_ih = (const float*)d_in[3];
    const float* b_ih = (const float*)d_in[4];
    const float* W_hh = (const float*)d_in[5];
    const float* b_hh = (const float*)d_in[6];
    const float* ln_w = (const float*)d_in[7];
    const float* ln_b = (const float*)d_in[8];

    // workspace layout: XH f16 (8MB) | Wcat f16 (16MB) | preact f32 (32MB)
    _Float16* XH   = (_Float16*)d_ws;
    _Float16* Wc   = XH + (size_t)BATCH * KTOT;
    float*    pre  = (float*)(Wc + (size_t)NTOT * KTOT);

    float* h_out = (float*)d_out;
    float* c_out = h_out + (size_t)BATCH * HID;

    // 1) convert/pack to f16 with K-concatenation
    int t1 = BATCH * HID / 4;
    pack_f16_kernel<<<t1 / 256, 256, 0, stream>>>(x, XH, HID, KTOT, 0,   t1);
    pack_f16_kernel<<<t1 / 256, 256, 0, stream>>>(h, XH, HID, KTOT, HID, t1);
    int t2 = NTOT * HID / 4;
    pack_f16_kernel<<<t2 / 256, 256, 0, stream>>>(W_ih, Wc, HID, KTOT, 0,   t2);
    pack_f16_kernel<<<t2 / 256, 256, 0, stream>>>(W_hh, Wc, HID, KTOT, HID, t2);

    // 2) WMMA GEMM -> preact
    dim3 grid(NTOT / BN, BATCH / BM);
    gemm_wmma_kernel<<<grid, 256, 0, stream>>>(XH, Wc, pre);

    // 3) gates + layernorm + outputs
    lstm_ln_kernel<<<BATCH, 256, 0, stream>>>(pre, c, b_ih, b_hh, ln_w, ln_b,
                                              h_out, c_out);
}
